// BitLinear_24670292148857
// MI455X (gfx1250) — 8-bit matrix operand order repaired and checked by reading, compiled, not run on hardware
//
#include <hip/hip_runtime.h>
#include <stdint.h>

typedef __attribute__((ext_vector_type(8))) int v8i;

#define DIN   2048
#define DOUT  2048
#define NROWS 16384
#define WN    (DIN * DOUT)

#define OFF_PARTIAL   0
#define OFF_SCL       2048
#define OFF_ROWSCALE  4096
#define OFF_XQ        131072ull
#define OFF_WQ        33685504ull

__global__ __launch_bounds__(256) void wabs_partial_k(const float4* __restrict__ w,
                                                      float* __restrict__ partial) {
    __shared__ float red[256];
    const int tid = threadIdx.x;
    float s = 0.f;
    for (int i = blockIdx.x * 256 + tid; i < WN / 4; i += 512 * 256) {
        float4 v = w[i];
        s += fabsf(v.x) + fabsf(v.y) + fabsf(v.z) + fabsf(v.w);
    }
    red[tid] = s;
    __syncthreads();
    for (int off = 128; off > 0; off >>= 1) {
        if (tid < off) red[tid] += red[tid + off];
        __syncthreads();
    }
    if (tid == 0) partial[blockIdx.x] = red[0];
}

__global__ __launch_bounds__(256) void wscale_final_k(const float* __restrict__ partial,
                                                      float* __restrict__ scl) {
    __shared__ float red[256];
    const int tid = threadIdx.x;
    red[tid] = partial[tid] + partial[tid + 256];
    __syncthreads();
    for (int off = 128; off > 0; off >>= 1) {
        if (tid < off) red[tid] += red[tid + off];
        __syncthreads();
    }
    if (tid == 0) {
        float scale = red[0] / (float)WN;
        scl[0] = scale;
        scl[1] = 1.0f / (scale + 1e-8f);
    }
}

__global__ __launch_bounds__(256) void wquant_k(const float4* __restrict__ w,
                                                const float* __restrict__ scl,
                                                char4* __restrict__ wq) {
    const int gid = blockIdx.x * 256 + threadIdx.x;
    const float inv = scl[1];
    float4 v = w[gid];
    char4 q;
    q.x = (signed char)fminf(fmaxf(rintf(v.x * inv), -1.f), 1.f);
    q.y = (signed char)fminf(fmaxf(rintf(v.y * inv), -1.f), 1.f);
    q.z = (signed char)fminf(fmaxf(rintf(v.z * inv), -1.f), 1.f);
    q.w = (signed char)fminf(fmaxf(rintf(v.w * inv), -1.f), 1.f);
    wq[gid] = q;
}

__global__ __launch_bounds__(256) void xquant_k(const float4* __restrict__ x,
                                                float* __restrict__ row_scale,
                                                uint2* __restrict__ xq) {
    __shared__ float red[256];
    __shared__ float sinv;
    const int tid = threadIdx.x;
    const int row = blockIdx.x;
    const float4* xr = x + (size_t)row * (DIN / 4);
    float4 a = xr[tid * 2];
    float4 b = xr[tid * 2 + 1];
    float m = fmaxf(fmaxf(fabsf(a.x), fabsf(a.y)), fmaxf(fabsf(a.z), fabsf(a.w)));
    m = fmaxf(m, fmaxf(fmaxf(fabsf(b.x), fabsf(b.y)), fmaxf(fabsf(b.z), fabsf(b.w))));
    red[tid] = m;
    __syncthreads();
    for (int off = 128; off > 0; off >>= 1) {
        if (tid < off) red[tid] = fmaxf(red[tid], red[tid + off]);
        __syncthreads();
    }
    if (tid == 0) {
        float g = fmaxf(red[0], 1e-8f);
        row_scale[row] = g * (1.0f / 127.0f);
        sinv = 127.0f / g;
    }
    __syncthreads();
    const float inv = sinv;
    float vals[8] = {a.x, a.y, a.z, a.w, b.x, b.y, b.z, b.w};
    int q[8];
#pragma unroll
    for (int i = 0; i < 8; ++i)
        q[i] = (int)fminf(fmaxf(rintf(vals[i] * inv), -128.f), 127.f);
    uint32_t lo = (q[0] & 0xff) | ((q[1] & 0xff) << 8) | ((q[2] & 0xff) << 16) | ((q[3] & 0xff) << 24);
    uint32_t hi = (q[4] & 0xff) | ((q[5] & 0xff) << 8) | ((q[6] & 0xff) << 16) | ((q[7] & 0xff) << 24);
    xq[(size_t)row * (DIN / 8) + tid] = make_uint2(lo, hi);
}

__device__ __forceinline__ v8i pack_a(uint64_t a0, uint64_t a1, uint64_t a2, uint64_t a3) {
    v8i v;
    v[0] = (int)(uint32_t)a0; v[1] = (int)(a0 >> 32);
    v[2] = (int)(uint32_t)a1; v[3] = (int)(a1 >> 32);
    v[4] = (int)(uint32_t)a2; v[5] = (int)(a2 >> 32);
    v[6] = (int)(uint32_t)a3; v[7] = (int)(a3 >> 32);
    return v;
}
__device__ __forceinline__ v8i pack_b(int4 lo, int4 hi) {
    v8i v;
    v[0] = lo.x; v[1] = lo.y; v[2] = lo.z; v[3] = lo.w;
    v[4] = hi.x; v[5] = hi.y; v[6] = hi.z; v[7] = hi.w;
    return v;
}

__global__ __launch_bounds__(256) void bitgemm_i8_k(
    const int8_t* __restrict__ xq, const int8_t* __restrict__ wq,
    const float* __restrict__ row_scale, const float* __restrict__ scl,
    const float* __restrict__ bias, float* __restrict__ out) {
    const int lane = threadIdx.x & 31;
    const int wav  = threadIdx.x >> 5;
    const int r    = lane & 15;
    const int h    = lane >> 4;
    const int m0   = (blockIdx.x * 8 + wav) << 4;
    const int n0   = blockIdx.y << 6;

    const uint64_t* A = (const uint64_t*)(xq + (size_t)(m0 + r) * DIN + (h << 4));
    const int4* B0 = (const int4*)(wq + (size_t)(n0 +  0 + r) * DIN + (h << 4));
    const int4* B1 = (const int4*)(wq + (size_t)(n0 + 16 + r) * DIN + (h << 4));
    const int4* B2 = (const int4*)(wq + (size_t)(n0 + 32 + r) * DIN + (h << 4));
    const int4* B3 = (const int4*)(wq + (size_t)(n0 + 48 + r) * DIN + (h << 4));

    v8i acc0 = {}, acc1 = {}, acc2 = {}, acc3 = {};

#pragma unroll 2
    for (int k0 = 0; k0 < DIN; k0 += 64) {
        const int ka = k0 >> 3;
        const int kb = k0 >> 4;
        uint64_t a0 = A[ka], a1 = A[ka + 1], a2 = A[ka + 4], a3 = A[ka + 5];
        int4 b0l = B0[kb], b0h = B0[kb + 2];
        int4 b1l = B1[kb], b1h = B1[kb + 2];
        int4 b2l = B2[kb], b2h = B2[kb + 2];
        int4 b3l = B3[kb], b3h = B3[kb + 2];
        v8i a = pack_a(a0, a1, a2, a3);
        acc0 = __builtin_amdgcn_wmma_i32_16x16x64_iu8(true, a, true, pack_b(b0l, b0h), acc0, false, false);
        acc1 = __builtin_amdgcn_wmma_i32_16x16x64_iu8(true, a, true, pack_b(b1l, b1h), acc1, false, false);
        acc2 = __builtin_amdgcn_wmma_i32_16x16x64_iu8(true, a, true, pack_b(b2l, b2h), acc2, false, false);
        acc3 = __builtin_amdgcn_wmma_i32_16x16x64_iu8(true, a, true, pack_b(b3l, b3h), acc3, false, false);
    }

    const float wsc = scl[0];
    float rs[8];
#pragma unroll
    for (int v = 0; v < 8; ++v)
        rs[v] = row_scale[m0 + v + 8 * h] * wsc;

    {
        const int n = n0 + r;
        const float bn = bias[n];
        float* op = out + (size_t)(m0 + 8 * h) * DOUT + n;
#pragma unroll
        for (int v = 0; v < 8; ++v)
            op[(size_t)v * DOUT] = bn + rs[v] * (float)acc0[v];
    }
    {
        const int n = n0 + 16 + r;
        const float bn = bias[n];
        float* op = out + (size_t)(m0 + 8 * h) * DOUT + n;
#pragma unroll
        for (int v = 0; v < 8; ++v)
            op[(size_t)v * DOUT] = bn + rs[v] * (float)acc1[v];
    }
    {
        const int n = n0 + 32 + r;
        const float bn = bias[n];
        float* op = out + (size_t)(m0 + 8 * h) * DOUT + n;
#pragma unroll
        for (int v = 0; v < 8; ++v)
            op[(size_t)v * DOUT] = bn + rs[v] * (float)acc2[v];
    }
    {
        const int n = n0 + 48 + r;
        const float bn = bias[n];
        float* op = out + (size_t)(m0 + 8 * h) * DOUT + n;
#pragma unroll
        for (int v = 0; v < 8; ++v)
            op[(size_t)v * DOUT] = bn + rs[v] * (float)acc3[v];
    }
}

extern "C" void kernel_launch(void* const* d_in, const int* in_sizes, int n_in,
                              void* d_out, int out_size, void* d_ws, size_t ws_size,
                              hipStream_t stream) {
    const float* x    = (const float*)d_in[0];
    const float* w    = (const float*)d_in[1];
    const float* bias = (const float*)d_in[2];
    float* out = (float*)d_out;

    char* ws = (char*)d_ws;
    float*  partial  = (float*)(ws + OFF_PARTIAL);
    float*  scl      = (float*)(ws + OFF_SCL);
    float*  rowscale = (float*)(ws + OFF_ROWSCALE);
    int8_t* xq       = (int8_t*)(ws + OFF_XQ);
    int8_t* wq       = (int8_t*)(ws + OFF_WQ);

    wabs_partial_k<<<512, 256, 0, stream>>>((const float4*)w, partial);
    wscale_final_k<<<1, 256, 0, stream>>>(partial, scl);
    wquant_k<<<WN / 4 / 256, 256, 0, stream>>>((const float4*)w, scl, (char4*)wq);
    xquant_k<<<NROWS, 256, 0, stream>>>((const float4*)x, rowscale, (uint2*)xq);
    bitgemm_i8_k<<<dim3(NROWS / 16 / 8, DOUT / 64), 256, 0, stream>>>(
        xq, wq, rowscale, scl, bias, out);
}
